// GraphFeatureExtractor_64896955842860
// MI455X (gfx1250) — matrix operand order repaired and checked by reading, not compiled, not run on hardware
//
#include <hip/hip_runtime.h>
#include <hip/hip_bf16.h>

typedef _Float16 f16;
typedef __attribute__((ext_vector_type(16))) _Float16 v16h;
typedef __attribute__((ext_vector_type(8)))  _Float16 v8h;
typedef __attribute__((ext_vector_type(8)))  float    v8f;
typedef __attribute__((ext_vector_type(4)))  int      v4i;

#define BB   8192
#define NN   36
#define FF   64
#define HH   64
#define RR   4
#define LL   3
#define NP   48          // padded node count (3 x 16)
#define OUTD 1024
#define KFC  4608        // 36 * 128

// Workspace layout (element = f16). Needs ~85.1 MB of d_ws.
constexpr size_t OFF_WM   = 0;                                   // [L][R][64][64]  W_msg^T
constexpr size_t OFF_WS   = OFF_WM + (size_t)LL * RR * 64 * 64;  // [L][64][64]     W_self^T
constexpr size_t OFF_WFC1 = OFF_WS + (size_t)LL * 64 * 64;       // [64][64]        W_fc1^T
constexpr size_t OFF_WFC2 = OFF_WFC1 + (size_t)64 * 64;          // [1024][4608]    W_fc2^T
constexpr size_t OFF_X    = OFF_WFC2 + (size_t)KFC * OUTD;       // [B][4608]       concat(h, feature)

static __device__ __forceinline__ float lrelu(float x) { return fmaxf(x, 0.01f * x); }

union V16u { v16h v; v8h h[2]; };

// A-operand (16x32 f16, MxK): lane holds row M=lane&15; lanes 0-15 K=k0..k0+15, lanes 16-31 K=k0+16..k0+31.
static __device__ __forceinline__ v16h loadA(const f16* base, int stride, int row0, int k0, int lane) {
    int r = row0 + (lane & 15);
    int c = k0 + 16 * (lane >> 4);
    const f16* p = base + r * stride + c;
    V16u t;
    t.h[0] = *(const v8h*)(p);
    t.h[1] = *(const v8h*)(p + 8);
    return t.v;
}

// B-operand (32x16 f16, KxN) read from B^T stored row-major [N][K]:
// lane holds column N=lane&15; lanes 0-15 K=k0..k0+15, lanes 16-31 K=k0+16..k0+31.
static __device__ __forceinline__ v16h loadB(const f16* base, int stride, int row0, int k0, int lane) {
    int r = row0 + (lane & 15);
    int c = k0 + 16 * (lane >> 4);
    const f16* p = base + r * stride + c;
    V16u t;
    t.h[0] = *(const v8h*)(p);
    t.h[1] = *(const v8h*)(p + 8);
    return t.v;
}

static __device__ __forceinline__ v8f wmma32f16(v16h a, v16h b, v8f c) {
    return __builtin_amdgcn_wmma_f32_16x16x32_f16(false, a, false, b, (short)0, c, false, false);
}

// ---- CDNA5 async global->LDS copy (ASYNCcnt-tracked), with sync fallback ----
#if defined(__HIP_DEVICE_COMPILE__) && __has_builtin(__builtin_amdgcn_global_load_async_to_lds_b128)
#define HAVE_ASYNC 1
typedef __attribute__((address_space(1))) v4i gv4i;   // global int4*
typedef __attribute__((address_space(3))) v4i lv4i;   // LDS int4*
#else
#define HAVE_ASYNC 0
#endif

static __device__ __forceinline__ void copy16_g2l(const f16* g, f16* l) {
#if HAVE_ASYNC
    __builtin_amdgcn_global_load_async_to_lds_b128((gv4i*)(void*)g, (lv4i*)(void*)l, 0, 0);
#else
    *(v8h*)l = *(const v8h*)g;
#endif
}
static __device__ __forceinline__ void async_wait_le3() {
#if HAVE_ASYNC
#if __has_builtin(__builtin_amdgcn_s_wait_asynccnt)
    __builtin_amdgcn_s_wait_asynccnt(3);
#else
    asm volatile("s_wait_asynccnt 3" ::: "memory");
#endif
#endif
}
static __device__ __forceinline__ void async_wait_0() {
#if HAVE_ASYNC
#if __has_builtin(__builtin_amdgcn_s_wait_asynccnt)
    __builtin_amdgcn_s_wait_asynccnt(0);
#else
    asm volatile("s_wait_asynccnt 0" ::: "memory");
#endif
#endif
}

// ---------------------------------------------------------------------------
// Kernel 0a: small weights -> f16 transposed
// ---------------------------------------------------------------------------
__global__ __launch_bounds__(256)
void prep_small_kernel(const float* __restrict__ Wmsg, const float* __restrict__ Wself,
                       const float* __restrict__ Wfc1, f16* __restrict__ ws) {
    int i = blockIdx.x * 256 + threadIdx.x;
    if (i < LL * RR * 64 * 64) {            // W_msg^T [l][r][h][f]
        int lr = i / 4096, rem = i % 4096, h = rem >> 6, f = rem & 63;
        ws[OFF_WM + i] = (f16)Wmsg[(size_t)lr * 4096 + f * 64 + h];
    }
    if (i < LL * 64 * 64) {                 // W_self^T [l][h][f]
        int l = i / 4096, rem = i % 4096, h = rem >> 6, f = rem & 63;
        ws[OFF_WS + i] = (f16)Wself[(size_t)l * 4096 + f * 64 + h];
    }
    if (i < 64 * 64) {                      // W_fc1^T [h][f]
        int h = i >> 6, f = i & 63;
        ws[OFF_WFC1 + i] = (f16)Wfc1[f * 64 + h];
    }
}

// ---------------------------------------------------------------------------
// Kernel 0b: W_fc2 [4608][1024] f32 -> W_fc2^T [1024][4608] f16 (tiled, coalesced)
// ---------------------------------------------------------------------------
__global__ __launch_bounds__(256)
void prep_fc2t_kernel(const float* __restrict__ W, f16* __restrict__ out) {
    __shared__ float t[32][33];
    const int k0 = blockIdx.x * 32;          // 144 blocks over K=4608
    const int n0 = blockIdx.y * 32;          // 32 blocks over N=1024
    const int tx = threadIdx.x & 31, ty = threadIdx.x >> 5;   // 32 x 8
#pragma unroll
    for (int i = 0; i < 32; i += 8)
        t[ty + i][tx] = W[(size_t)(k0 + ty + i) * OUTD + n0 + tx];
    __syncthreads();
#pragma unroll
    for (int i = 0; i < 32; i += 8)
        out[(size_t)(n0 + ty + i) * KFC + k0 + tx] = (f16)t[tx][ty + i];
}

// ---------------------------------------------------------------------------
// Kernel 1: per-graph 3 RGNN layers + skip branch.  One block (6 waves) / graph.
// ---------------------------------------------------------------------------
__global__ __launch_bounds__(192)
void gnn_layers_kernel(const float* __restrict__ feat,
                       const float* __restrict__ Wgate,  // [L][R][64]
                       const float* __restrict__ bself,  // [L][64]
                       const float* __restrict__ bfc1,   // [64]
                       const int*   __restrict__ adj,    // [B][R][36][36]
                       const f16*   __restrict__ wmT,    // [L][R][64][64]
                       const f16*   __restrict__ wsT,    // [L][64][64]
                       const f16*   __restrict__ wfc1T,  // [64][64]
                       f16* __restrict__ X)              // [B][4608]
{
    // LDS: 64,000 bytes total
    __shared__ f16 lds[NP * 64 + RR * 64 * 64 + RR * NP * 64 + RR * 64];
    f16* hb   = lds;                       // [48][64]  node features (f16; pad rows may hold junk)
    f16* gm   = hb + NP * 64;              // [R][64][64] gated-msg^T: [h][s] (s cols 48..63 stay 0)
    f16* aadj = gm + RR * 64 * 64;         // [R][48][64] adjacency^T: [d][s] f16 (pads 0)
    f16* gate = aadj + RR * NP * 64;       // [R][64]    (pads 0 forever -> kills junk in h pads)

    const int tid  = threadIdx.x;
    const int lane = tid & 31;
    const int wave = tid >> 5;             // 0..5
    const int b    = blockIdx.x;

    // zero all LDS (establishes every pad)
    {
        v8h z = {};
        const int tot8 = (NP * 64 + RR * 64 * 64 + RR * NP * 64 + RR * 64) / 8;
        for (int j = tid; j < tot8; j += 192) *(v8h*)(lds + (size_t)j * 8) = z;
    }
    __syncthreads();

    // feat -> hb (f16);  adj -> aadj transposed (A^T[d][s])
    const float* fb = feat + (size_t)b * NN * FF;
    for (int j = tid; j < NN * FF; j += 192) hb[j] = (f16)fb[j];
    const int* ab = adj + (size_t)b * RR * NN * NN;
    for (int j = tid; j < RR * NN * NN; j += 192) {
        int r = j / (NN * NN);
        int rem = j - r * NN * NN;
        int s = rem / NN, d = rem - s * NN;
        aadj[(r * NP + d) * 64 + s] = (f16)ab[j];
    }
    __syncthreads();

    // ---- skip branch: lrelu(feat @ W_fc1 + b_fc1) -> X[b][m][64..127]
    for (int j = wave; j < 12; j += 6) {
        int mt = j >> 2, nt = j & 3;
        v8f acc = {};
#pragma unroll
        for (int k0 = 0; k0 < 64; k0 += 32) {
            v16h a = loadA(hb, 64, mt * 16, k0, lane);
            v16h w = loadB(wfc1T, 64, nt * 16, k0, lane);
            acc = wmma32f16(a, w, acc);
        }
        int n  = nt * 16 + (lane & 15);
        int m0 = mt * 16 + 8 * (lane >> 4);
        float bias = bfc1[n];
#pragma unroll
        for (int v = 0; v < 8; ++v)
            gm[(m0 + v) * 64 + n] = (f16)lrelu(acc[v] + bias);   // stage row-major in gm
    }
    __syncthreads();
    f16* Xb = X + (size_t)b * KFC;
    for (int j = tid; j < NN * 8; j += 192) {
        int m = j >> 3, c = (j & 7) * 8;
        *(v8h*)(Xb + m * 128 + 64 + c) = *(const v8h*)(gm + m * 64 + c);
    }
    __syncthreads();
    {   // re-zero staged region of gm (pads must be zero for agg K=64)
        v8h z = {};
        for (int j = tid; j < NP * 64 / 8; j += 192) *(v8h*)(gm + (size_t)j * 8) = z;
    }
    __syncthreads();

    // ---- layers
    for (int l = 0; l < LL; ++l) {
        // per-(r,node) scalar gates (VALU; 144 dots of length 64)
        if (tid < RR * NN) {
            int r = tid / NN, n2 = tid - r * NN;
            const float* wg = Wgate + (size_t)(l * RR + r) * FF;
            const f16* hr = hb + n2 * 64;
            float acc = 0.f;
#pragma unroll 8
            for (int f = 0; f < FF; ++f) acc += (float)hr[f] * wg[f];
            gate[r * 64 + n2] = (f16)(1.f / (1.f + __expf(-acc)));
        }
        __syncthreads();

        // msg = h @ W_msg[r]; gmsg^T[h][s] = msg[s][h]*gate[r][s] -> gm
        for (int j = wave; j < 48; j += 6) {
            int r = j / 12, t = j - r * 12, mt = t >> 2, nt = t & 3;
            const f16* wb = wmT + (size_t)(l * RR + r) * 4096;
            v8f acc = {};
#pragma unroll
            for (int k0 = 0; k0 < 64; k0 += 32) {
                v16h a = loadA(hb, 64, mt * 16, k0, lane);
                v16h w = loadB(wb, 64, nt * 16, k0, lane);
                acc = wmma32f16(a, w, acc);
            }
            int n  = nt * 16 + (lane & 15);
            int s0 = mt * 16 + 8 * (lane >> 4);
            v8h g = *(const v8h*)(gate + r * 64 + s0);   // gate==0 on pad rows
            v8h o;
#pragma unroll
            for (int v = 0; v < 8; ++v) o[v] = (f16)(acc[v] * (float)g[v]);
            *(v8h*)(gm + (r * 64 + n) * 64 + s0) = o;    // one contiguous b128: transposed store
        }
        __syncthreads();

        // agg = sum_r A_r^T @ gmsg_r  +  h @ W_self  + b_self ; lrelu -> hb (register-staged)
        v8h res[2];
        int resoff[2];
#pragma unroll
        for (int jj = 0; jj < 2; ++jj) {
            int j = wave + jj * 6;
            int mt = j >> 2, nt = j & 3;
            v8f acc = {};
#pragma unroll
            for (int r = 0; r < RR; ++r) {
#pragma unroll
                for (int k0 = 0; k0 < 64; k0 += 32) {
                    v16h a = loadA(aadj + r * NP * 64, 64, mt * 16, k0, lane);
                    v16h g = loadB(gm + r * 64 * 64, 64, nt * 16, k0, lane);
                    acc = wmma32f16(a, g, acc);
                }
            }
            const f16* wsb = wsT + (size_t)l * 4096;
#pragma unroll
            for (int k0 = 0; k0 < 64; k0 += 32) {
                v16h a = loadA(hb, 64, mt * 16, k0, lane);
                v16h w = loadB(wsb, 64, nt * 16, k0, lane);
                acc = wmma32f16(a, w, acc);
            }
            int n  = nt * 16 + (lane & 15);
            int m0 = mt * 16 + 8 * (lane >> 4);
            float bias = bself[l * HH + n];
#pragma unroll
            for (int v = 0; v < 8; ++v) res[jj][v] = (f16)lrelu(acc[v] + bias);
            resoff[jj] = m0 * 64 + n;
        }
        __syncthreads();   // all reads of hb done before overwrite
#pragma unroll
        for (int jj = 0; jj < 2; ++jj)
#pragma unroll
            for (int v = 0; v < 8; ++v)
                hb[resoff[jj] + v * 64] = res[jj][v];    // unguarded: pad rows neutralized by gate==0
        __syncthreads();
    }

    // final h -> X[b][m][0..63] (coalesced b128; reads only valid rows)
    for (int j = tid; j < NN * 8; j += 192) {
        int m = j >> 3, c = (j & 7) * 8;
        *(v8h*)(Xb + m * 128 + c) = *(const v8h*)(hb + m * 64 + c);
    }
}

// ---------------------------------------------------------------------------
// Kernel 2: out = lrelu(X[B][4608] @ W_fc2 + b_fc2) -> f32 [B][1024]
// 128x64 tile / 256-thread block, double-buffered LDS, async global->LDS fills.
// ---------------------------------------------------------------------------
__global__ __launch_bounds__(256)
void fc2_kernel(const f16* __restrict__ X,      // [B][4608]
                const f16* __restrict__ WT,     // [1024][4608]
                const float* __restrict__ bias, // [1024]
                float* __restrict__ out)        // [B][1024]
{
    __shared__ f16 Xs[2][128 * 40];   // 32 cols + 8 pad
    __shared__ f16 Ws[2][64 * 40];

    const int tid = threadIdx.x, lane = tid & 31, wave = tid >> 5;
    const size_t rowX0 = (size_t)blockIdx.x * 128;
    const size_t rowW0 = (size_t)blockIdx.y * 64;

    const int xr = tid >> 1;          // 0..127
    const int xc = (tid & 1) * 16;    // 0 / 16
    const int wr = tid >> 2;          // 0..63
    const int wc = (tid & 3) * 8;     // 0,8,16,24

    const f16* Xg = X  + (rowX0 + xr) * KFC + xc;
    const f16* Wg = WT + (rowW0 + wr) * KFC + wc;
    __builtin_prefetch(Wg, 0, 1);     // global_prefetch_b8

    v8f acc[4] = {{}, {}, {}, {}};
    const int NK = KFC / 32;          // 144

    // prologue fill: chunk 0 -> buffer 0 (3 async ops per thread)
    copy16_g2l(Xg,     &Xs[0][xr * 40 + xc]);
    copy16_g2l(Xg + 8, &Xs[0][xr * 40 + xc + 8]);
    copy16_g2l(Wg,     &Ws[0][wr * 40 + wc]);

    for (int kt = 0; kt < NK; ++kt) {
        int cur = kt & 1;
        if (kt + 1 < NK) {            // fill next buffer, then wait only for our previous fill
            int k0 = (kt + 1) * 32;
            copy16_g2l(Xg + k0,     &Xs[cur ^ 1][xr * 40 + xc]);
            copy16_g2l(Xg + k0 + 8, &Xs[cur ^ 1][xr * 40 + xc + 8]);
            copy16_g2l(Wg + k0,     &Ws[cur ^ 1][wr * 40 + wc]);
            async_wait_le3();         // async loads complete in order
        } else {
            async_wait_0();
        }
        __syncthreads();              // all waves' fills of `cur` visible

        v16h a = loadA(&Xs[cur][wave * 16 * 40], 40, 0, 0, lane);
#pragma unroll
        for (int nt = 0; nt < 4; ++nt) {
            v16h w = loadB(&Ws[cur][0], 40, nt * 16, 0, lane);
            acc[nt] = wmma32f16(a, w, acc[nt]);
        }
        __syncthreads();              // WAR: buffer `cur` reusable next iteration
    }

    const int m0 = 8 * (lane >> 4);
    const size_t orow = rowX0 + wave * 16 + m0;
#pragma unroll
    for (int nt = 0; nt < 4; ++nt) {
        int col = (int)rowW0 + nt * 16 + (lane & 15);
        float bv = bias[col];
#pragma unroll
        for (int v = 0; v < 8; ++v)
            out[(orow + v) * OUTD + col] = lrelu(acc[nt][v] + bv);
    }
}

// ---------------------------------------------------------------------------
extern "C" void kernel_launch(void* const* d_in, const int* in_sizes, int n_in,
                              void* d_out, int out_size, void* d_ws, size_t ws_size,
                              hipStream_t stream) {
    (void)in_sizes; (void)n_in; (void)out_size; (void)ws_size;
    const float* feat  = (const float*)d_in[0];
    const float* Wmsg  = (const float*)d_in[1];
    const float* Wgate = (const float*)d_in[2];
    const float* Wself = (const float*)d_in[3];
    const float* bself = (const float*)d_in[4];
    const float* Wfc1  = (const float*)d_in[5];
    const float* bfc1  = (const float*)d_in[6];
    const float* Wfc2  = (const float*)d_in[7];
    const float* bfc2  = (const float*)d_in[8];
    const int*   adjp  = (const int*)d_in[9];

    f16* ws    = (f16*)d_ws;
    f16* wmT   = ws + OFF_WM;
    f16* wsT   = ws + OFF_WS;
    f16* wfc1T = ws + OFF_WFC1;
    f16* wfc2T = ws + OFF_WFC2;
    f16* Xp    = ws + OFF_X;

    prep_small_kernel<<<(LL * RR * 64 * 64 + 255) / 256, 256, 0, stream>>>(Wmsg, Wself, Wfc1, ws);
    prep_fc2t_kernel<<<dim3(KFC / 32, OUTD / 32), 256, 0, stream>>>(Wfc2, wfc2T);
    gnn_layers_kernel<<<BB, 192, 0, stream>>>(feat, Wgate, bself, bfc1, adjp,
                                              wmT, wsT, wfc1T, Xp);
    fc2_kernel<<<dim3(BB / 128, OUTD / 64), 256, 0, stream>>>(Xp, wfc2T, bfc2,
                                                              (float*)d_out);
}
